// MyViT_57080115363902
// MI455X (gfx1250) — compile-verified, matrix operand order checked by reading, not run on hardware
//
#include <hip/hip_runtime.h>
#include <hip/hip_bf16.h>
#include <math.h>

// ---------------- types ----------------
typedef __bf16 bf16;
typedef float  v8f   __attribute__((ext_vector_type(8)));
typedef bf16   v16bf __attribute__((ext_vector_type(16)));
typedef unsigned int u32x4 __attribute__((ext_vector_type(4)));
typedef int          i32x4 __attribute__((ext_vector_type(4)));

union Frag { v16bf v; u32x4 q[2]; };

__device__ __forceinline__ u32x4 zero4() { u32x4 v; v[0]=0u; v[1]=0u; v[2]=0u; v[3]=0u; return v; }
__device__ __forceinline__ void fragZero(Frag& f) { f.q[0] = zero4(); f.q[1] = zero4(); }

// CDNA5 async global->LDS DMA (ASYNCcnt-tracked), guarded so the build
// falls back to load+ds_store if the toolchain lacks the builtins.
#if __has_builtin(__builtin_amdgcn_global_load_async_to_lds_b128) && \
    __has_builtin(__builtin_amdgcn_s_wait_asynccnt)
#define USE_ASYNC_LDS 1
typedef __attribute__((address_space(1))) i32x4 g_i32x4;   // global (AS1) v4i
typedef __attribute__((address_space(3))) i32x4 l_i32x4;   // LDS (AS3) v4i
__device__ __forceinline__ void async_b128(const void* gsrc, void* ldst)
{
    // generic->AS1: same bits; generic->AS3: LDS offset lives in addr[31:0] (ISA 10.2)
    __builtin_amdgcn_global_load_async_to_lds_b128(
        (g_i32x4*)(unsigned long long)gsrc,
        (l_i32x4*)(unsigned int)(unsigned long long)ldst, 0, 0);
}
#endif

// ---------------- model dims ----------------
#define VIT_BATCH  32
#define VIT_SEQ    197
#define VIT_HID    768
#define VIT_HEADS  12
#define VIT_DH     64
#define VIT_BLOCKS 12
#define VIT_MLP    3072
#define VIT_MROWS  (VIT_BATCH*VIT_SEQ)   /* 6304 */
#define VIT_PROWS  (VIT_BATCH*196)       /* 6272 */
#define VIT_OUTD   1000

// flags for GEMM
#define GF_OUT_BF16 1
#define GF_GELU     2

// =====================================================================
// Generic bf16 WMMA GEMM: C[M,N] = A[M,K] * B[K,N] + bias, opt. GELU.
// A bf16 row-major (lda), B bf16 row-major (ldb) transposed into LDS,
// C f32 or bf16 (ldc). blockIdx.z batches via *OffZ element offsets.
// Block = 128 threads (4 waves); 64x64 tile; wave w -> rows [16w,16w+16).
// Tile loads are UNPREDICATED: out-of-range rows/cols only feed acc
// lanes that the (predicated) epilogue discards, and the small
// over-reads stay inside the workspace carve. A tile goes through the
// CDNA5 async DMA path (GLOBAL_LOAD_ASYNC_TO_LDS_B128).
// =====================================================================
__launch_bounds__(128)
__global__ void gemm_wmma(const bf16* __restrict__ A, int lda, long aOffZ,
                          const bf16* __restrict__ B, int ldb, long bOffZ,
                          const float* __restrict__ bias, long biasOffZ,
                          void* __restrict__ Cv, int ldc, long cOffZ,
                          int M, int N, int K, int flags)
{
    __shared__ __align__(16) bf16 sA[64][32];
    __shared__ __align__(16) bf16 sB[64][32];   // transposed: sB[n][k]

    const int z = blockIdx.z;
    A += (long)z * aOffZ;
    B += (long)z * bOffZ;
    const float* bz = bias ? (bias + (long)z * biasOffZ) : nullptr;

    const int m0 = blockIdx.y * 64, n0 = blockIdx.x * 64;
    const int tid = threadIdx.x, wave = tid >> 5, lane = tid & 31;
    const int ln16 = lane & 15, half = lane >> 4, b0 = half ? 8 : 0;

    v8f acc[4];
#pragma unroll
    for (int nt = 0; nt < 4; nt++)
#pragma unroll
        for (int j = 0; j < 8; j++) acc[nt][j] = 0.f;

    for (int k0 = 0; k0 < K; k0 += 32) {
        // ---- A tile: 64 rows x 32 cols, 16B chunks ----
#pragma unroll
        for (int i = 0; i < 2; i++) {
            int ch = tid * 2 + i;
            int r = ch >> 2, kc = (ch & 3) * 8;
            const bf16* src = A + (long)(m0 + r) * lda + k0 + kc;
#ifdef USE_ASYNC_LDS
            async_b128(src, &sA[r][kc]);
#else
            *(u32x4*)&sA[r][kc] = *(const u32x4*)src;
#endif
        }
        // ---- B tile [32k x 64n] -> transposed sB[n][k] ----
        // vectorized along contiguous N, scalar scatter into LDS
#pragma unroll
        for (int i = 0; i < 2; i++) {
            int job = tid * 2 + i;          // 0..255
            int kk = job >> 3;              // 0..31
            int n8 = (job & 7) * 8;         // 0..56 step 8
            union { u32x4 q; bf16 e[8]; } bl;
            bl.q = *(const u32x4*)(B + (long)(k0 + kk) * ldb + n0 + n8);
#pragma unroll
            for (int t = 0; t < 8; t++) sB[n8 + t][kk] = bl.e[t];
        }
#ifdef USE_ASYNC_LDS
        __builtin_amdgcn_s_wait_asynccnt(0);
#endif
        __syncthreads();

        Frag fa;
        fa.q[0] = *(const u32x4*)&sA[wave * 16 + ln16][b0];
        fa.q[1] = *(const u32x4*)&sA[wave * 16 + ln16][b0 + 16];
#pragma unroll
        for (int nt = 0; nt < 4; nt++) {
            Frag fb;
            fb.q[0] = *(const u32x4*)&sB[nt * 16 + ln16][b0];
            fb.q[1] = *(const u32x4*)&sB[nt * 16 + ln16][b0 + 16];
            acc[nt] = __builtin_amdgcn_wmma_f32_16x16x32_bf16(
                false, fa.v, false, fb.v, (short)0, acc[nt], false, false);
        }
        __syncthreads();
    }

    // epilogue: bias (+GELU), predicated store
#pragma unroll
    for (int nt = 0; nt < 4; nt++) {
        int gn = n0 + nt * 16 + ln16;
        float bvv = (bz && gn < N) ? bz[gn] : 0.f;
#pragma unroll
        for (int j = 0; j < 8; j++) {
            int gm = m0 + wave * 16 + j + half * 8;
            if (gm < M && gn < N) {
                float v = acc[nt][j] + bvv;
                if (flags & GF_GELU) v = 0.5f * v * (1.f + erff(v * 0.70710678118f));
                long off = (long)z * cOffZ + (long)gm * ldc + gn;
                if (flags & GF_OUT_BF16) ((bf16*)Cv)[off] = (bf16)v;
                else                     ((float*)Cv)[off] = v;
            }
        }
    }
}

// =====================================================================
// Flash attention per (batch, head). Q/K/V bf16 laid out [H][B*SEQ][64].
// Seq padded to 224 (14 tiles). 8 waves, each owns whole q-row-tiles.
// Scores: B^T = K rows -> K fragments loaded directly from global.
// P tiles re-laid out C->A via per-wave LDS staging; V kept transposed
// in LDS for the P*V WMMAs. Online softmax via 16-lane shfl reductions.
// =====================================================================
__launch_bounds__(256)
__global__ void attn_flash(const bf16* __restrict__ Q, const bf16* __restrict__ K,
                           const bf16* __restrict__ V, float* __restrict__ O,
                           float scale)
{
    __shared__ __align__(16) bf16 vT[64][224];     // V^T, 28.7 KB
    __shared__ __align__(16) bf16 pst[8][16 * 32]; // per-wave P staging, 8 KB

    const int bh = blockIdx.x;
    const int b = bh / VIT_HEADS, hh = bh % VIT_HEADS;
    const long base = ((long)hh * VIT_MROWS + (long)b * VIT_SEQ) * VIT_DH;
    const int tid = threadIdx.x, wave = tid >> 5, lane = tid & 31;
    const int ln16 = lane & 15, half = lane >> 4, b0 = half ? 8 : 0;

    // build V^T in LDS (zero padded rows)
    for (int idx = tid; idx < 224 * 64; idx += 256) {
        int s = idx >> 6, e = idx & 63;
        bf16 val = (bf16)0.f;
        if (s < VIT_SEQ) val = V[base + (long)s * VIT_DH + e];
        vT[e][s] = val;
    }
    __syncthreads();

    for (int qt = wave; qt < 14; qt += 8) {
        const int qs = qt * 16 + ln16;
        Frag qf0, qf1;
        if (qs < VIT_SEQ) {
            const bf16* qr = Q + base + (long)qs * VIT_DH;
            qf0.q[0] = *(const u32x4*)(qr + b0);
            qf0.q[1] = *(const u32x4*)(qr + b0 + 16);
            qf1.q[0] = *(const u32x4*)(qr + 32 + b0);
            qf1.q[1] = *(const u32x4*)(qr + 32 + b0 + 16);
        } else { fragZero(qf0); fragZero(qf1); }

        float mrow[8], lrow[8];
        v8f oacc[4];
#pragma unroll
        for (int j = 0; j < 8; j++) { mrow[j] = -1e30f; lrow[j] = 0.f; }
#pragma unroll
        for (int nt = 0; nt < 4; nt++)
#pragma unroll
            for (int j = 0; j < 8; j++) oacc[nt][j] = 0.f;

        for (int kp = 0; kp < 7; kp++) {
#pragma unroll
            for (int jj = 0; jj < 2; jj++) {
                const int kt = kp * 2 + jj;
                const int ks = kt * 16 + ln16;
                Frag kf0, kf1;
                if (ks < VIT_SEQ) {
                    const bf16* kr = K + base + (long)ks * VIT_DH;
                    kf0.q[0] = *(const u32x4*)(kr + b0);
                    kf0.q[1] = *(const u32x4*)(kr + b0 + 16);
                    kf1.q[0] = *(const u32x4*)(kr + 32 + b0);
                    kf1.q[1] = *(const u32x4*)(kr + 32 + b0 + 16);
                } else { fragZero(kf0); fragZero(kf1); }

                v8f sc;
#pragma unroll
                for (int j = 0; j < 8; j++) sc[j] = 0.f;
                sc = __builtin_amdgcn_wmma_f32_16x16x32_bf16(
                    false, qf0.v, false, kf0.v, (short)0, sc, false, false);
                sc = __builtin_amdgcn_wmma_f32_16x16x32_bf16(
                    false, qf1.v, false, kf1.v, (short)0, sc, false, false);

                const bool colok = (ks < VIT_SEQ);  // column index == ks (lane-dependent)
                float pv[8], corr[8];
#pragma unroll
                for (int j = 0; j < 8; j++) {
                    float sv = colok ? sc[j] * scale : -1e30f;
                    float rm = sv;  // row max across the 16 lanes holding this row
                    rm = fmaxf(rm, __shfl_xor(rm, 1));
                    rm = fmaxf(rm, __shfl_xor(rm, 2));
                    rm = fmaxf(rm, __shfl_xor(rm, 4));
                    rm = fmaxf(rm, __shfl_xor(rm, 8));
                    float nm = fmaxf(mrow[j], rm);
                    corr[j] = __expf(mrow[j] - nm);
                    float pe = __expf(sv - nm);
                    float ps = pe;
                    ps += __shfl_xor(ps, 1);
                    ps += __shfl_xor(ps, 2);
                    ps += __shfl_xor(ps, 4);
                    ps += __shfl_xor(ps, 8);
                    lrow[j] = lrow[j] * corr[j] + ps;
                    mrow[j] = nm;
                    pv[j] = pe;
                }
#pragma unroll
                for (int nt = 0; nt < 4; nt++)
#pragma unroll
                    for (int j = 0; j < 8; j++) oacc[nt][j] *= corr[j];
                // stage P tile (C layout -> A layout in LDS)
#pragma unroll
                for (int j = 0; j < 8; j++) {
                    int r = j + half * 8;
                    pst[wave][r * 32 + jj * 16 + ln16] = (bf16)pv[j];
                }
            }
            asm volatile("s_wait_dscnt 0" ::: "memory");
            Frag pf;
            pf.q[0] = *(const u32x4*)&pst[wave][ln16 * 32 + b0];
            pf.q[1] = *(const u32x4*)&pst[wave][ln16 * 32 + b0 + 16];
#pragma unroll
            for (int nt = 0; nt < 4; nt++) {
                const int e = nt * 16 + ln16;
                Frag vf;
                vf.q[0] = *(const u32x4*)&vT[e][kp * 32 + b0];
                vf.q[1] = *(const u32x4*)&vT[e][kp * 32 + b0 + 16];
                oacc[nt] = __builtin_amdgcn_wmma_f32_16x16x32_bf16(
                    false, pf.v, false, vf.v, (short)0, oacc[nt], false, false);
            }
        }
        // finalize: divide by l, write O[b][s][hh*64+e]
#pragma unroll
        for (int nt = 0; nt < 4; nt++) {
            const int e = nt * 16 + ln16;
#pragma unroll
            for (int j = 0; j < 8; j++) {
                const int s = qt * 16 + j + half * 8;
                if (s < VIT_SEQ)
                    O[((long)b * VIT_SEQ + s) * VIT_HID + hh * VIT_DH + e] =
                        oacc[nt][j] / lrow[j];
            }
        }
    }
}

// ---------------- LayerNorm (row of 768) -> bf16 ----------------
__launch_bounds__(256)
__global__ void ln_to_bf16(const float* __restrict__ X, const float* __restrict__ g,
                           const float* __restrict__ bb, bf16* __restrict__ out)
{
    const int r = blockIdx.x, tid = threadIdx.x;
    const float v0 = X[(long)r * 768 + tid];
    const float v1 = X[(long)r * 768 + tid + 256];
    const float v2 = X[(long)r * 768 + tid + 512];
    float s = v0 + v1 + v2, q = v0 * v0 + v1 * v1 + v2 * v2;
    for (int o = 16; o; o >>= 1) { s += __shfl_down(s, o); q += __shfl_down(q, o); }
    __shared__ float rs[8], rq[8];
    const int wave = tid >> 5, lane = tid & 31;
    if (!lane) { rs[wave] = s; rq[wave] = q; }
    __syncthreads();
    s = 0.f; q = 0.f;
    for (int w = 0; w < 8; w++) { s += rs[w]; q += rq[w]; }
    const float mean = s * (1.f / 768.f);
    const float var = q * (1.f / 768.f) - mean * mean;
    const float inv = rsqrtf(var + 1e-5f);
    out[(long)r * 768 + tid]       = (bf16)((v0 - mean) * inv * g[tid]       + bb[tid]);
    out[(long)r * 768 + tid + 256] = (bf16)((v1 - mean) * inv * g[tid + 256] + bb[tid + 256]);
    out[(long)r * 768 + tid + 512] = (bf16)((v2 - mean) * inv * g[tid + 512] + bb[tid + 512]);
}

// ---------------- small utility kernels ----------------
__global__ void f32_to_bf16_k(const float* __restrict__ s, bf16* __restrict__ d, long n)
{
    for (long i = (long)blockIdx.x * blockDim.x + threadIdx.x; i < n;
         i += (long)gridDim.x * blockDim.x)
        d[i] = (bf16)s[i];
}

__global__ void patchify_bf16(const float* __restrict__ x, bf16* __restrict__ xp, long total)
{
    for (long idx = (long)blockIdx.x * blockDim.x + threadIdx.x; idx < total;
         idx += (long)gridDim.x * blockDim.x) {
        long n = idx / (196L * 768);
        int rem = (int)(idx % (196L * 768));
        int p = rem / 768, col = rem % 768;
        int py = p / 14, px = p % 14;
        int c = col >> 8, r2 = col & 255;
        int iy = r2 >> 4, ix = r2 & 15;
        long src = ((n * 3 + c) * 224 + (py * 16 + iy)) * 224L + (px * 16 + ix);
        xp[idx] = (bf16)x[src];
    }
}

__global__ void assemble_h(const float* __restrict__ tok, const float* __restrict__ cls,
                           const float* __restrict__ pe, float* __restrict__ h, long total)
{
    for (long idx = (long)blockIdx.x * blockDim.x + threadIdx.x; idx < total;
         idx += (long)gridDim.x * blockDim.x) {
        long b = idx / (197L * 768);
        int rem = (int)(idx % (197L * 768));
        int s = rem / 768, c = rem % 768;
        float v = (s == 0) ? cls[c] : tok[(b * 196 + s - 1) * 768L + c];
        h[idx] = v + pe[(long)s * 768 + c];
    }
}

__global__ void resid_add(float* __restrict__ h, const float* __restrict__ a,
                          const float* __restrict__ m, long n)
{
    for (long i = (long)blockIdx.x * blockDim.x + threadIdx.x; i < n;
         i += (long)gridDim.x * blockDim.x)
        h[i] = h[i] + a[i] + m[i];
}

__global__ void extract_cls(const float* __restrict__ h, bf16* __restrict__ clsb)
{
    int idx = blockIdx.x * 256 + threadIdx.x;  // 32*768
    int b = idx / 768, c = idx % 768;
    clsb[idx] = (bf16)h[((long)b * VIT_SEQ) * 768 + c];
}

__launch_bounds__(256)
__global__ void softmax1000(const float* __restrict__ logits, float* __restrict__ out)
{
    const int b = blockIdx.x, tid = threadIdx.x;
    const float* row = logits + (long)b * VIT_OUTD;
    float m = -1e30f;
    for (int j = tid; j < VIT_OUTD; j += 256) m = fmaxf(m, row[j]);
    for (int o = 16; o; o >>= 1) m = fmaxf(m, __shfl_xor(m, o));
    __shared__ float rm[8], rs[8];
    const int wave = tid >> 5;
    if (!(tid & 31)) rm[wave] = m;
    __syncthreads();
    m = rm[0];
    for (int w = 1; w < 8; w++) m = fmaxf(m, rm[w]);
    float s = 0.f;
    for (int j = tid; j < VIT_OUTD; j += 256) s += __expf(row[j] - m);
    for (int o = 16; o; o >>= 1) s += __shfl_xor(s, o);
    if (!(tid & 31)) rs[wave] = s;
    __syncthreads();
    s = 0.f;
    for (int w = 0; w < 8; w++) s += rs[w];
    const float invs = 1.f / s;
    for (int j = tid; j < VIT_OUTD; j += 256)
        out[(long)b * VIT_OUTD + j] = __expf(row[j] - m) * invs;
}

// =====================================================================
// host orchestration
// =====================================================================
extern "C" void kernel_launch(void* const* d_in, const int* in_sizes, int n_in,
                              void* d_out, int out_size, void* d_ws, size_t ws_size,
                              hipStream_t stream)
{
    (void)in_sizes; (void)n_in; (void)out_size; (void)ws_size;

    const float* x       = (const float*)d_in[0];
    const float* W_map   = (const float*)d_in[1];
    const float* b_map   = (const float*)d_in[2];
    const float* cls_tok = (const float*)d_in[3];
    const float* pos_emb = (const float*)d_in[4];
    const float* ln1_s   = (const float*)d_in[5];
    const float* ln1_b   = (const float*)d_in[6];
    const float* Wq      = (const float*)d_in[7];
    const float* bq      = (const float*)d_in[8];
    const float* Wk      = (const float*)d_in[9];
    const float* bk      = (const float*)d_in[10];
    const float* Wv      = (const float*)d_in[11];
    const float* bv      = (const float*)d_in[12];
    const float* ln2_s   = (const float*)d_in[13];
    const float* ln2_b   = (const float*)d_in[14];
    const float* W1      = (const float*)d_in[15];
    const float* b1      = (const float*)d_in[16];
    const float* W2      = (const float*)d_in[17];
    const float* b2      = (const float*)d_in[18];
    const float* W_out   = (const float*)d_in[19];
    const float* b_out   = (const float*)d_in[20];
    float* out = (float*)d_out;

    // ---- workspace carve (~170 MB) ----
    char* p = (char*)d_ws;
    auto alloc = [&](size_t bytes) -> char* {
        char* r = p; p += (bytes + 255) & ~(size_t)255; return r;
    };
    float* h    = (float*)alloc((size_t)VIT_MROWS * 768 * 4);
    bf16* xln1  = (bf16*)alloc((size_t)VIT_MROWS * 768 * 2);
    bf16* xln2  = (bf16*)alloc((size_t)VIT_MROWS * 768 * 2);
    bf16* qb    = (bf16*)alloc((size_t)VIT_HEADS * VIT_MROWS * 64 * 2);
    bf16* kb    = (bf16*)alloc((size_t)VIT_HEADS * VIT_MROWS * 64 * 2);
    bf16* vb    = (bf16*)alloc((size_t)VIT_HEADS * VIT_MROWS * 64 * 2);
    float* attn = (float*)alloc((size_t)VIT_MROWS * 768 * 4);
    bf16* m1    = (bf16*)alloc((size_t)VIT_MROWS * 3072 * 2);
    float* m2   = (float*)alloc((size_t)VIT_MROWS * 768 * 4);
    bf16* wmapb = (bf16*)alloc((size_t)768 * 768 * 2);
    bf16* wqb   = (bf16*)alloc((size_t)12 * 12 * 64 * 64 * 2);
    bf16* wkb   = (bf16*)alloc((size_t)12 * 12 * 64 * 64 * 2);
    bf16* wvb   = (bf16*)alloc((size_t)12 * 12 * 64 * 64 * 2);
    bf16* w1b   = (bf16*)alloc((size_t)768 * 3072 * 2);
    bf16* w2b   = (bf16*)alloc((size_t)768 * 3072 * 2);
    bf16* woutb = (bf16*)alloc((size_t)768 * 1000 * 2);
    bf16* clsb  = (bf16*)alloc((size_t)32 * 768 * 2);
    float* logit= (float*)alloc((size_t)32 * 1000 * 4);
    // pre-loop aliases (safe: consumed before the block loop overwrites them)
    float* tok = attn;   // [6272,768] f32
    bf16*  xp  = m1;     // [6272,768] bf16

    const dim3 B128(128), B256(256);
    auto cgrid = [](long n) { long g = (n + 255) / 256; if (g > 65535) g = 65535; return dim3((unsigned)g); };

    // ---- one-time weight conversions ----
    f32_to_bf16_k<<<cgrid(768L * 768), B256, 0, stream>>>(W_map, wmapb, 768L * 768);
    f32_to_bf16_k<<<cgrid(12L * 12 * 64 * 64), B256, 0, stream>>>(Wq, wqb, 12L * 12 * 64 * 64);
    f32_to_bf16_k<<<cgrid(12L * 12 * 64 * 64), B256, 0, stream>>>(Wk, wkb, 12L * 12 * 64 * 64);
    f32_to_bf16_k<<<cgrid(12L * 12 * 64 * 64), B256, 0, stream>>>(Wv, wvb, 12L * 12 * 64 * 64);
    f32_to_bf16_k<<<cgrid(768L * 1000), B256, 0, stream>>>(W_out, woutb, 768L * 1000);

    // ---- patch embed: tok = patchify(x) @ W_map + b_map ----
    patchify_bf16<<<cgrid((long)VIT_PROWS * 768), B256, 0, stream>>>(x, xp, (long)VIT_PROWS * 768);
    gemm_wmma<<<dim3(768 / 64, VIT_PROWS / 64, 1), B128, 0, stream>>>(
        xp, 768, 0, wmapb, 768, 0, b_map, 0, tok, 768, 0, VIT_PROWS, 768, 768, 0);
    assemble_h<<<cgrid((long)VIT_MROWS * 768), B256, 0, stream>>>(
        tok, cls_tok, pos_emb, h, (long)VIT_MROWS * 768);

    const float scale = 0.28867513459481287f;  // 1/sqrt(N_HEADS=12), per reference
    const int mt = (VIT_MROWS + 63) / 64;      // 99

    for (int bl = 0; bl < VIT_BLOCKS; bl++) {
        ln_to_bf16<<<dim3(VIT_MROWS), B256, 0, stream>>>(h, ln1_s + bl * 768, ln1_b + bl * 768, xln1);
        ln_to_bf16<<<dim3(VIT_MROWS), B256, 0, stream>>>(h, ln2_s + bl * 768, ln2_b + bl * 768, xln2);

        // per-head QKV projections (z = head; A column-shifted by 64 per head)
        const dim3 qg(1, mt, VIT_HEADS);
        gemm_wmma<<<qg, B128, 0, stream>>>(
            xln1, 768, 64, wqb + (size_t)bl * 12 * 64 * 64, 64, 64 * 64,
            bq + (size_t)bl * 12 * 64, 64, qb, 64, (long)VIT_MROWS * 64,
            VIT_MROWS, 64, 64, GF_OUT_BF16);
        gemm_wmma<<<qg, B128, 0, stream>>>(
            xln1, 768, 64, wkb + (size_t)bl * 12 * 64 * 64, 64, 64 * 64,
            bk + (size_t)bl * 12 * 64, 64, kb, 64, (long)VIT_MROWS * 64,
            VIT_MROWS, 64, 64, GF_OUT_BF16);
        gemm_wmma<<<qg, B128, 0, stream>>>(
            xln1, 768, 64, wvb + (size_t)bl * 12 * 64 * 64, 64, 64 * 64,
            bv + (size_t)bl * 12 * 64, 64, vb, 64, (long)VIT_MROWS * 64,
            VIT_MROWS, 64, 64, GF_OUT_BF16);

        attn_flash<<<dim3(VIT_BATCH * VIT_HEADS), B256, 0, stream>>>(qb, kb, vb, attn, scale);

        // MLP: m2 = gelu(xln2 @ W1 + b1) @ W2 + b2
        f32_to_bf16_k<<<cgrid(768L * 3072), B256, 0, stream>>>(
            W1 + (size_t)bl * 768 * 3072, w1b, 768L * 3072);
        gemm_wmma<<<dim3(3072 / 64, mt, 1), B128, 0, stream>>>(
            xln2, 768, 0, w1b, 3072, 0, b1 + (size_t)bl * 3072, 0,
            m1, 3072, 0, VIT_MROWS, 3072, 768, GF_OUT_BF16 | GF_GELU);
        f32_to_bf16_k<<<cgrid(768L * 3072), B256, 0, stream>>>(
            W2 + (size_t)bl * 3072 * 768, w2b, 768L * 3072);
        gemm_wmma<<<dim3(768 / 64, mt, 1), B128, 0, stream>>>(
            m1, 3072, 0, w2b, 768, 0, b2 + (size_t)bl * 768, 0,
            m2, 768, 0, VIT_MROWS, 768, 3072, 0);

        // h = h + attn + mlp  (both branches read the original h per reference)
        resid_add<<<cgrid((long)VIT_MROWS * 768), B256, 0, stream>>>(
            h, attn, m2, (long)VIT_MROWS * 768);
    }

    // head: softmax(cls @ W_out + b_out)
    extract_cls<<<dim3(32 * 768 / 256), B256, 0, stream>>>(h, clsb);
    gemm_wmma<<<dim3((VIT_OUTD + 63) / 64, 1, 1), B128, 0, stream>>>(
        clsb, 768, 0, woutb, 1000, 0, b_out, 0, logit, 1000, 0, 32, VIT_OUTD, 768, 0);
    softmax1000<<<dim3(VIT_BATCH), B256, 0, stream>>>(logit, out);
}
